// GNNModel_21345987461277
// MI455X (gfx1250) — matrix operand order repaired and checked by reading, not compiled, not run on hardware
//
#include <hip/hip_runtime.h>
#include <hip/hip_bf16.h>
#include <math.h>

// ---------------- problem constants (match reference) ----------------
#define N_NODE  100000
#define N_USER  20000
#define HDIM    128
#define BSZ     512
#define SLEN    20
#define NT      (BSZ*SLEN)       // 10240 session nodes
#define NE      (2*NT)           // 20480 session edges
#define EGE     1000000          // global-graph edges
#define TOTAL   (N_NODE+N_USER)  // 120000

#define CDIV(a,b) (((a)+(b)-1)/(b))

typedef __attribute__((ext_vector_type(16))) _Float16 v16h;
typedef __attribute__((ext_vector_type(8)))  float    v8f;

__device__ __forceinline__ float sigmoidf_(float x){ return 1.0f/(1.0f+__expf(-x)); }

// ====================================================================
// WMMA GEMM:  C[M,N] = act( A0[M,K0] * W0 + A1[M,K1] * W1 + bias0 + bias1 )
//   TRANSB=0:  W element (k,n) at W[k*ldW + n]
//   TRANSB=1:  W element (k,n) at W[n*ldW + k]   (for X @ W^T)
//   ACT: 0 = none, 1 = sigmoid
// One wave32 computes a 64x16 C super-tile (4 M-tiles): the B fragment is
// loaded once per K-step and reused across 4 v_wmma_f32_16x16x32_f16 ops
// (f32 accumulate), quartering B-side L2 traffic. M % 64 == 0, N % 16 == 0,
// K % 32 == 0 for all shapes in this model.
// ====================================================================
#define MT 4   // M-tiles per wave

// A fragment 16x32 (MxK), per ISA 7.12.2: lanes 0-15 M=0..15 (K groups
// 0..7 / 16..23), lanes 16-31 same M with K groups +8.
__device__ __forceinline__ v16h load_a_frag(const float* __restrict__ A, int lda,
                                            int row0, int kbase, int lane) {
  v16h a;
  const float* base = A + (size_t)(row0 + (lane & 15)) * (size_t)lda + kbase;
  const int hoff = (lane >= 16) ? 8 : 0;
#pragma unroll
  for (int g = 0; g < 8; ++g) {
    int k = ((g < 4) ? (2*g) : (8 + 2*g)) + hoff;  // 0,2,4,6,16,18,20,22 (+hoff)
    a[2*g]   = (_Float16)base[k];
    a[2*g+1] = (_Float16)base[k+1];
  }
  return a;
}

// B fragment 32x16 (KxN): lane -> column N=lane&15; same K order as A
// (lanes 0-15: K 0..7,16..23; lanes 16-31: K 8..15,24..31).
template<int TRANSB>
__device__ __forceinline__ v16h load_b_frag(const float* __restrict__ W, int ldW,
                                            int kbase, int col0, int lane) {
  v16h b;
  const int n  = col0 + (lane & 15);
  const int k0 = kbase + ((lane >= 16) ? 8 : 0);
#pragma unroll
  for (int h = 0; h < 16; ++h) {
    const int kk = k0 + ((h < 8) ? h : (h + 8));
    float v = TRANSB ? W[(size_t)n * (size_t)ldW + kk]
                     : W[(size_t)kk * (size_t)ldW + n];
    b[h] = (_Float16)v;
  }
  return b;
}

template<int TRANSB, int ACT>
__global__ void wmma_gemm(const float* __restrict__ A0, const float* __restrict__ W0, int ldW0, int K0,
                          const float* __restrict__ A1, const float* __restrict__ W1, int ldW1, int K1,
                          const float* __restrict__ bias0, const float* __restrict__ bias1,
                          float* __restrict__ C, int M, int N) {
  const int lane  = threadIdx.x & 31;
  const int wid   = (blockIdx.x * blockDim.x + threadIdx.x) >> 5;
  const int ntile = N >> 4;
  const int T     = (M >> 6) * ntile;          // super-tiles of 64 rows
  if (wid >= T) return;                        // uniform per-wave: EXEC stays all-1s
  const int tm = wid / ntile;                  // super-tile row index
  const int tn = wid - tm * ntile;
  const int row0 = tm * (16 * MT);

  v8f c[MT];
#pragma unroll
  for (int m = 0; m < MT; ++m) c[m] = (v8f){0.f,0.f,0.f,0.f,0.f,0.f,0.f,0.f};

  for (int kb = 0; kb < K0; kb += 32) {
    v16h b = load_b_frag<TRANSB>(W0, ldW0, kb, tn*16, lane);
#pragma unroll
    for (int m = 0; m < MT; ++m) {
      v16h a = load_a_frag(A0, K0, row0 + m*16, kb, lane);
      c[m] = __builtin_amdgcn_wmma_f32_16x16x32_f16(false, a, false, b, (short)0, c[m], false, false);
    }
  }
  if (A1) {
    for (int kb = 0; kb < K1; kb += 32) {
      v16h b = load_b_frag<TRANSB>(W1, ldW1, kb, tn*16, lane);
#pragma unroll
      for (int m = 0; m < MT; ++m) {
        v16h a = load_a_frag(A1, K1, row0 + m*16, kb, lane);
        c[m] = __builtin_amdgcn_wmma_f32_16x16x32_f16(false, a, false, b, (short)0, c[m], false, false);
      }
    }
  }

  // C/D layout: n = lane&15, rows mb..mb+7 where mb = (lane>=16 ? 8 : 0)
  const int n = tn*16 + (lane & 15);
  float badd = 0.f;
  if (bias0) badd += bias0[n];
  if (bias1) badd += bias1[n];
#pragma unroll
  for (int m = 0; m < MT; ++m) {
    const int mb = row0 + m*16 + ((lane >= 16) ? 8 : 0);
#pragma unroll
    for (int v = 0; v < 8; ++v) {
      float val = c[m][v] + badd;
      if (ACT == 1) val = sigmoidf_(val);
      C[(size_t)(mb + v) * (size_t)N + n] = val;
    }
  }
}

// ====================================================================
// Elementwise / scatter kernels
// ====================================================================
__global__ void k_zero(float* __restrict__ p, long long n) {
  long long t = (long long)blockIdx.x * blockDim.x + threadIdx.x;
  if (t < n) p[t] = 0.f;
}

// out[i,:] = tab[idx[i]+off, :]
__global__ void k_gather(const int* __restrict__ idx, int off,
                         const float* __restrict__ tab, float* __restrict__ out, int nrows) {
  int t = blockIdx.x * blockDim.x + threadIdx.x;
  if (t >= nrows * HDIM) return;
  int i = t >> 7, h = t & 127;
  out[t] = tab[(size_t)(idx[i] + off) * HDIM + h];
}

// out[i,:] = 0.5*(a[i,:] + tab[idx[i]+off, :])
__global__ void k_avg_gather(const float* __restrict__ a, const int* __restrict__ idx, int off,
                             const float* __restrict__ tab, float* __restrict__ out, int nrows) {
  int t = blockIdx.x * blockDim.x + threadIdx.x;
  if (t >= nrows * HDIM) return;
  int i = t >> 7, h = t & 127;
  out[t] = 0.5f * (a[t] + tab[(size_t)(idx[i] + off) * HDIM + h]);
}

// GGNN edge propagation (atomic segment-sum over session edges)
__global__ void k_ggnn_edges(const int* __restrict__ ei, const float* __restrict__ ec,
                             const float* __restrict__ din, const float* __restrict__ dout_,
                             const float* __restrict__ t_in, const float* __restrict__ t_out,
                             float* __restrict__ m_in, float* __restrict__ m_out) {
  int t = blockIdx.x * blockDim.x + threadIdx.x;
  if (t >= NE * HDIM) return;
  int e = t >> 7, h = t & 127;
  int s = ei[e], d = ei[NE + e];
  atomicAdd(&m_in [(size_t)d * HDIM + h], ec[e] * din [e] * t_in [(size_t)s * HDIM + h]);
  atomicAdd(&m_out[(size_t)s * HDIM + h], ec[e] * dout_[e] * t_out[(size_t)d * HDIM + h]);
}

// GRU elementwise combine: gi/gh are [NT, 3H]
__global__ void k_gru(const float* __restrict__ gi, const float* __restrict__ gh,
                      const float* __restrict__ emb, float* __restrict__ hid) {
  int t = blockIdx.x * blockDim.x + threadIdx.x;
  if (t >= NT * HDIM) return;
  int i = t >> 7, h = t & 127;
  const float* gir = gi + (size_t)i * (3*HDIM);
  const float* ghr = gh + (size_t)i * (3*HDIM);
  float r  = sigmoidf_(gir[h]         + ghr[h]);
  float z  = sigmoidf_(gir[HDIM + h]  + ghr[HDIM + h]);
  float nn = tanhf    (gir[2*HDIM + h] + r * ghr[2*HDIM + h]);
  hid[t] = (1.f - z) * nn + z * emb[t];
}

// last[b] = cumsum(bincount(batch))[b] - 1   (single block of BSZ threads)
__global__ void k_session_last(const int* __restrict__ batch, int nt, int* __restrict__ last) {
  __shared__ int cnt[BSZ];
  int t = threadIdx.x;
  cnt[t] = 0;
  __syncthreads();
  for (int i = t; i < nt; i += BSZ) atomicAdd(&cnt[batch[i]], 1);
  __syncthreads();
  for (int off = 1; off < BSZ; off <<= 1) {
    int v = cnt[t];
    int add = (t >= off) ? cnt[t - off] : 0;
    __syncthreads();
    cnt[t] = v + add;
    __syncthreads();
  }
  last[t] = cnt[t] - 1;
}

// alpha[i] = a[i,:] . qw + qb
__global__ void k_rowdot(const float* __restrict__ a, const float* __restrict__ qw,
                         const float* __restrict__ qb, float* __restrict__ alpha, int nrows) {
  int i = blockIdx.x * blockDim.x + threadIdx.x;
  if (i >= nrows) return;
  float s = qb[0];
  const float* row = a + (size_t)i * HDIM;
  for (int h = 0; h < HDIM; ++h) s += row[h] * qw[h];
  alpha[i] = s;
}

// s_g[batch[i],:] += num_count[i]*alpha[i]*h[i,:]
__global__ void k_i2s_scatter(const int* __restrict__ batch, const float* __restrict__ numc,
                              const float* __restrict__ alpha, const float* __restrict__ hm,
                              float* __restrict__ sg) {
  int t = blockIdx.x * blockDim.x + threadIdx.x;
  if (t >= NT * HDIM) return;
  int i = t >> 7, h = t & 127;
  atomicAdd(&sg[(size_t)batch[i] * HDIM + h], numc[i] * alpha[i] * hm[t]);
}

// SAGE: neighbor count
__global__ void k_sage_cnt(const int* __restrict__ ei, float* __restrict__ cnt) {
  int e = blockIdx.x * blockDim.x + threadIdx.x;
  if (e >= EGE) return;
  atomicAdd(&cnt[ei[EGE + e]], 1.0f);
}

// SAGE: agg[dst,:] += x[src,:]  (4 floats per thread, 32 threads per edge)
__global__ void k_sage_agg(const int* __restrict__ ei, const float* __restrict__ x,
                           float* __restrict__ agg) {
  long long t = (long long)blockIdx.x * blockDim.x + threadIdx.x;
  if (t >= (long long)EGE * 32) return;
  int e  = (int)(t >> 5);
  int h0 = ((int)t & 31) * 4;
  int s = ei[e], d = ei[EGE + e];
  const float* xs = x   + (size_t)s * HDIM + h0;
  float*       ad = agg + (size_t)d * HDIM + h0;
#pragma unroll
  for (int j = 0; j < 4; ++j) atomicAdd(&ad[j], xs[j]);
}

__global__ void k_meandiv(float* __restrict__ agg, const float* __restrict__ cnt, int nrows) {
  int t = blockIdx.x * blockDim.x + threadIdx.x;
  if (t >= nrows * HDIM) return;
  agg[t] /= fmaxf(cnt[t >> 7], 1.0f);
}

// g = sigmoid([sess|user] . gw + gb); wuser = g*user + (1-g)*sess
__global__ void k_gate_combine(const float* __restrict__ sess, const float* __restrict__ user,
                               const float* __restrict__ gw, const float* __restrict__ gb,
                               float* __restrict__ wuser) {
  int b = blockIdx.x * blockDim.x + threadIdx.x;
  if (b >= BSZ) return;
  float s = gb[0];
  for (int h = 0; h < HDIM; ++h)
    s += sess[(size_t)b*HDIM + h] * gw[h] + user[(size_t)b*HDIM + h] * gw[HDIM + h];
  float g = sigmoidf_(s);
  for (int h = 0; h < HDIM; ++h)
    wuser[(size_t)b*HDIM + h] = g * user[(size_t)b*HDIM + h] + (1.f - g) * sess[(size_t)b*HDIM + h];
}

// alpha = sigmoid([gsess|guser].sw + sb); hfin = sess + alpha*gsess + (1-alpha)*guser
__global__ void k_suc_final(const float* __restrict__ sess, const float* __restrict__ gsess,
                            const float* __restrict__ guser, const float* __restrict__ sw,
                            const float* __restrict__ sb, float* __restrict__ hfin) {
  int b = blockIdx.x * blockDim.x + threadIdx.x;
  if (b >= BSZ) return;
  float s = sb[0];
  for (int h = 0; h < HDIM; ++h)
    s += gsess[(size_t)b*HDIM + h] * sw[h] + guser[(size_t)b*HDIM + h] * sw[HDIM + h];
  float a = sigmoidf_(s);
  for (int h = 0; h < HDIM; ++h)
    hfin[(size_t)b*HDIM + h] = sess[(size_t)b*HDIM + h]
                             + a * gsess[(size_t)b*HDIM + h]
                             + (1.f - a) * guser[(size_t)b*HDIM + h];
}

// pos/neg logits + targets into tail of d_out
__global__ void k_posneg(const float* __restrict__ user, const float* __restrict__ guser,
                         float* __restrict__ outp, float* __restrict__ tgt) {
  int b = blockIdx.x * blockDim.x + threadIdx.x;
  if (b >= BSZ) return;
  int br = (b + BSZ - 1) & (BSZ - 1);      // roll by 1
  float p = 0.f, ng = 0.f;
  for (int h = 0; h < HDIM; ++h) {
    float u = user[(size_t)b*HDIM + h];
    p  += u * guser[(size_t)b *HDIM + h];
    ng += u * guser[(size_t)br*HDIM + h];
  }
  outp[b]       = p;   outp[BSZ + b] = ng;
  tgt [b]       = 1.f; tgt [BSZ + b] = 0.f;
}

// ====================================================================
// Host-side orchestration
// ====================================================================
static inline int gemm_grid(int M, int N) {           // 256 threads = 8 waves/block
  int tiles = (M / (16*MT)) * (N / 16);               // 64-row super-tiles
  return CDIV(tiles, 8);
}

extern "C" void kernel_launch(void* const* d_in, const int* in_sizes, int n_in,
                              void* d_out, int out_size, void* d_ws, size_t ws_size,
                              hipStream_t stream) {
  // ---- inputs (setup_inputs dict order; params flattened in insertion order) ----
  const int*   x         = (const int*)  d_in[0];
  const int*   edge_idx  = (const int*)  d_in[1];   // [2, NE]
  const int*   batch     = (const int*)  d_in[2];
  const float* edge_cnt  = (const float*)d_in[3];
  const float* indeg     = (const float*)d_in[4];
  const float* outdeg    = (const float*)d_in[5];
  const float* num_count = (const float*)d_in[6];
  const int*   user_ids  = (const int*)  d_in[7];
  const int*   ei_i2u    = (const int*)  d_in[8];   // [2, EGE]
  const int*   ei_i2i    = (const int*)  d_in[9];
  const float* emb       = (const float*)d_in[10];  // [TOTAL, H]
  const float* ggnn_w_in = (const float*)d_in[11];
  const float* ggnn_w_out= (const float*)d_in[12];
  const float* gru_w_ih  = (const float*)d_in[13];  // [3H, 2H]
  const float* gru_w_hh  = (const float*)d_in[14];  // [3H, H]
  const float* gru_b_ih  = (const float*)d_in[15];
  const float* gru_b_hh  = (const float*)d_in[16];
  const float* i2u_wl    = (const float*)d_in[17];
  const float* i2u_wr    = (const float*)d_in[18];
  const float* i2u_b     = (const float*)d_in[19];
  const float* i2i_wl    = (const float*)d_in[20];
  const float* i2i_wr    = (const float*)d_in[21];
  const float* i2i_b     = (const float*)d_in[22];
  const float* s1_w1 = (const float*)d_in[23]; const float* s1_b1 = (const float*)d_in[24];
  const float* s1_w2 = (const float*)d_in[25]; const float* s1_b2 = (const float*)d_in[26];
  const float* s1_qw = (const float*)d_in[27]; const float* s1_qb = (const float*)d_in[28];
  const float* s1_w3 = (const float*)d_in[29]; const float* s1_b3 = (const float*)d_in[30];
  const float* s2_w1 = (const float*)d_in[31]; const float* s2_b1 = (const float*)d_in[32];
  const float* s2_w2 = (const float*)d_in[33]; const float* s2_b2 = (const float*)d_in[34];
  const float* s2_qw = (const float*)d_in[35]; const float* s2_qb = (const float*)d_in[36];
  const float* s2_w3 = (const float*)d_in[37]; const float* s2_b3 = (const float*)d_in[38];
  const float* gate_w = (const float*)d_in[39]; const float* gate_b = (const float*)d_in[40];
  const float* suc_w  = (const float*)d_in[41]; const float* suc_b  = (const float*)d_in[42];

  // ---- workspace bump allocator ----
  char* wsp = (char*)d_ws; size_t off = 0;
  auto alloc = [&](size_t nfloats) -> float* {
    float* p = (float*)(wsp + off);
    off += nfloats * sizeof(float);
    return p;
  };
  float* item_emb = alloc((size_t)NT*HDIM);
  float* t_in     = alloc((size_t)NT*HDIM);
  float* t_out    = alloc((size_t)NT*HDIM);
  float* m_in     = alloc((size_t)NT*HDIM);
  float* m_out    = alloc((size_t)NT*HDIM);
  float* gi       = alloc((size_t)NT*3*HDIM);
  float* gh       = alloc((size_t)NT*3*HDIM);
  float* hidden   = alloc((size_t)NT*HDIM);
  float* v_n      = alloc((size_t)BSZ*HDIM);
  float* vnr      = alloc((size_t)NT*HDIM);
  float* a_mat    = alloc((size_t)NT*HDIM);
  float* alpha    = alloc((size_t)NT);
  float* s_g      = alloc((size_t)BSZ*HDIM);
  float* sess_emb = alloc((size_t)BSZ*HDIM);
  float* user_emb = alloc((size_t)BSZ*HDIM);
  float* wuser    = alloc((size_t)BSZ*HDIM);
  float* gnn_item = alloc((size_t)NT*HDIM);
  float* gnn_user = alloc((size_t)BSZ*HDIM);
  float* gnn_sess = alloc((size_t)BSZ*HDIM);
  float* hfin     = alloc((size_t)BSZ*HDIM);
  float* aggbuf   = alloc((size_t)TOTAL*HDIM);
  float* cntbuf   = alloc((size_t)TOTAL);
  float* g1       = alloc((size_t)TOTAL*HDIM);
  float* g2       = alloc((size_t)TOTAL*HDIM);
  int*   last     = (int*)alloc((size_t)BSZ);

  float* scores   = (float*)d_out;                       // [BSZ, N_NODE]
  float* out_log  = scores + (size_t)BSZ * N_NODE;       // [2B]
  float* tgt      = out_log + 2*BSZ;                     // [2B]

  const int TB = 256;

  // ---- 1) item embedding gather (x is 1-based) ----
  k_gather<<<CDIV(NT*HDIM,TB), TB, 0, stream>>>(x, -1, emb, item_emb, NT);

  // ---- 2) GGNN: linears + weighted edge scatter ----
  wmma_gemm<0,0><<<gemm_grid(NT,HDIM), TB, 0, stream>>>(
      item_emb, ggnn_w_in, HDIM, HDIM, nullptr, nullptr, 0, 0, nullptr, nullptr, t_in, NT, HDIM);
  wmma_gemm<0,0><<<gemm_grid(NT,HDIM), TB, 0, stream>>>(
      item_emb, ggnn_w_out, HDIM, HDIM, nullptr, nullptr, 0, 0, nullptr, nullptr, t_out, NT, HDIM);
  k_zero<<<CDIV((long long)2*NT*HDIM,TB), TB, 0, stream>>>(m_in, (long long)2*NT*HDIM); // m_in+m_out contiguous
  k_ggnn_edges<<<CDIV(NE*HDIM,TB), TB, 0, stream>>>(edge_idx, edge_cnt, indeg, outdeg,
                                                    t_in, t_out, m_in, m_out);

  // ---- 3) GRU: gi = [m_in|m_out] @ w_ih^T + b_ih ; gh = item_emb @ w_hh^T + b_hh ----
  wmma_gemm<1,0><<<gemm_grid(NT,3*HDIM), TB, 0, stream>>>(
      m_in,  gru_w_ih,        2*HDIM, HDIM,
      m_out, gru_w_ih + HDIM, 2*HDIM, HDIM,
      gru_b_ih, nullptr, gi, NT, 3*HDIM);
  wmma_gemm<1,0><<<gemm_grid(NT,3*HDIM), TB, 0, stream>>>(
      item_emb, gru_w_hh, HDIM, HDIM, nullptr, nullptr, 0, 0,
      gru_b_hh, nullptr, gh, NT, 3*HDIM);
  k_gru<<<CDIV(NT*HDIM,TB), TB, 0, stream>>>(gi, gh, item_emb, hidden);

  // ---- 4) session pooling #1 (s1_) ----
  k_session_last<<<1, BSZ, 0, stream>>>(batch, NT, last);
  k_gather<<<CDIV(BSZ*HDIM,TB), TB, 0, stream>>>(last, 0, hidden, v_n, BSZ);
  k_gather<<<CDIV(NT*HDIM,TB), TB, 0, stream>>>(batch, 0, v_n, vnr, NT);
  wmma_gemm<0,1><<<gemm_grid(NT,HDIM), TB, 0, stream>>>(
      vnr, s1_w1, HDIM, HDIM, hidden, s1_w2, HDIM, HDIM, s1_b1, s1_b2, a_mat, NT, HDIM);
  k_rowdot<<<CDIV(NT,TB), TB, 0, stream>>>(a_mat, s1_qw, s1_qb, alpha, NT);
  k_zero<<<CDIV((long long)BSZ*HDIM,TB), TB, 0, stream>>>(s_g, (long long)BSZ*HDIM);
  k_i2s_scatter<<<CDIV(NT*HDIM,TB), TB, 0, stream>>>(batch, num_count, alpha, hidden, s_g);
  wmma_gemm<0,0><<<gemm_grid(BSZ,HDIM), TB, 0, stream>>>(
      v_n, s1_w3, HDIM, HDIM, s_g, s1_w3 + (size_t)HDIM*HDIM, HDIM, HDIM,
      s1_b3, nullptr, sess_emb, BSZ, HDIM);

  // ---- 5) user gate ----
  k_gather<<<CDIV(BSZ*HDIM,TB), TB, 0, stream>>>(user_ids, N_NODE, emb, user_emb, BSZ);
  k_gate_combine<<<CDIV(BSZ,TB), TB, 0, stream>>>(sess_emb, user_emb, gate_w, gate_b, wuser);

  // ---- 6) SAGE conv 1 (i2u) on full embedding table ----
  k_zero<<<CDIV((long long)TOTAL*HDIM + TOTAL, TB), TB, 0, stream>>>(aggbuf, (long long)TOTAL*HDIM + TOTAL);
  k_sage_cnt<<<CDIV(EGE,TB), TB, 0, stream>>>(ei_i2u, cntbuf);
  k_sage_agg<<<(int)CDIV((long long)EGE*32,TB), TB, 0, stream>>>(ei_i2u, emb, aggbuf);
  k_meandiv<<<CDIV(TOTAL*HDIM,TB), TB, 0, stream>>>(aggbuf, cntbuf, TOTAL);
  wmma_gemm<0,0><<<gemm_grid(TOTAL,HDIM), TB, 0, stream>>>(
      aggbuf, i2u_wl, HDIM, HDIM, emb, i2u_wr, HDIM, HDIM, i2u_b, nullptr, g1, TOTAL, HDIM);

  // ---- 7) SAGE conv 2 (i2i) ----
  k_zero<<<CDIV((long long)TOTAL*HDIM + TOTAL, TB), TB, 0, stream>>>(aggbuf, (long long)TOTAL*HDIM + TOTAL);
  k_sage_cnt<<<CDIV(EGE,TB), TB, 0, stream>>>(ei_i2i, cntbuf);
  k_sage_agg<<<(int)CDIV((long long)EGE*32,TB), TB, 0, stream>>>(ei_i2i, g1, aggbuf);
  k_meandiv<<<CDIV(TOTAL*HDIM,TB), TB, 0, stream>>>(aggbuf, cntbuf, TOTAL);
  wmma_gemm<0,0><<<gemm_grid(TOTAL,HDIM), TB, 0, stream>>>(
      aggbuf, i2i_wl, HDIM, HDIM, g1, i2i_wr, HDIM, HDIM, i2i_b, nullptr, g2, TOTAL, HDIM);

  // ---- 8) fuse with GNN output ----
  k_avg_gather<<<CDIV(NT*HDIM,TB), TB, 0, stream>>>(item_emb, x, -1, g2, gnn_item, NT);
  k_avg_gather<<<CDIV(BSZ*HDIM,TB), TB, 0, stream>>>(wuser, user_ids, N_NODE, g2, gnn_user, BSZ);

  // ---- 9) session pooling #2 (s2_) on gnn_item (reuse scratch) ----
  k_gather<<<CDIV(BSZ*HDIM,TB), TB, 0, stream>>>(last, 0, gnn_item, v_n, BSZ);
  k_gather<<<CDIV(NT*HDIM,TB), TB, 0, stream>>>(batch, 0, v_n, vnr, NT);
  wmma_gemm<0,1><<<gemm_grid(NT,HDIM), TB, 0, stream>>>(
      vnr, s2_w1, HDIM, HDIM, gnn_item, s2_w2, HDIM, HDIM, s2_b1, s2_b2, a_mat, NT, HDIM);
  k_rowdot<<<CDIV(NT,TB), TB, 0, stream>>>(a_mat, s2_qw, s2_qb, alpha, NT);
  k_zero<<<CDIV((long long)BSZ*HDIM,TB), TB, 0, stream>>>(s_g, (long long)BSZ*HDIM);
  k_i2s_scatter<<<CDIV(NT*HDIM,TB), TB, 0, stream>>>(batch, num_count, alpha, gnn_item, s_g);
  wmma_gemm<0,0><<<gemm_grid(BSZ,HDIM), TB, 0, stream>>>(
      v_n, s2_w3, HDIM, HDIM, s_g, s2_w3 + (size_t)HDIM*HDIM, HDIM, HDIM,
      s2_b3, nullptr, gnn_sess, BSZ, HDIM);

  // ---- 10) final combine + scores + pos/neg ----
  k_suc_final<<<CDIV(BSZ,TB), TB, 0, stream>>>(sess_emb, gnn_sess, gnn_user, suc_w, suc_b, hfin);
  // scores = hfin @ emb[:N_NODE]^T   (big one: 512 x 100000 x 128)
  wmma_gemm<1,0><<<gemm_grid(BSZ,N_NODE), TB, 0, stream>>>(
      hfin, emb, HDIM, HDIM, nullptr, nullptr, 0, 0, nullptr, nullptr, scores, BSZ, N_NODE);
  k_posneg<<<CDIV(BSZ,TB), TB, 0, stream>>>(user_emb, gnn_user, out_log, tgt);
}
